// MultiHeadAttention_83236466196860
// MI455X (gfx1250) — matrix operand order repaired and checked by reading, not compiled, not run on hardware
//
#include <hip/hip_runtime.h>

typedef __attribute__((ext_vector_type(16))) _Float16 v16h;
typedef __attribute__((ext_vector_type(8)))  _Float16 hvec8;
typedef __attribute__((ext_vector_type(8)))  float    v8f;

namespace {

constexpr int kB = 4, kT = 2048, kE = 512, kH = 8, kHS = 64;
constexpr int kC = kH * kHS;           // 512 concat-head dim
constexpr int kRowTiles = kT / 16;     // 128
constexpr float kLog2e = 1.44269504088896340736f;

// ---- fragment loaders (wave32, 16x16x32 f16 WMMA) -------------------------
// A (16x32, MxK) from row-major [rows][ld] at (row0,col0)=base:
//   lane half h, m=lane&15: VGPRs 0-3 hold K=16h..16h+7, VGPRs 4-7 hold K=16h+8..
__device__ __forceinline__ v16h load_a_frag(const _Float16* base, int ld) {
  const int lane = threadIdx.x & 31;
  const int hh = lane >> 4, m = lane & 15;
  const _Float16* r = base + (size_t)m * ld;
  hvec8 lo = *(const hvec8*)(r + 16 * hh);
  hvec8 hi = *(const hvec8*)(r + 16 * hh + 8);
  v16h a;
#pragma unroll
  for (int i = 0; i < 8; ++i) { a[i] = lo[i]; a[i + 8] = hi[i]; }
  return a;
}

// B (32x16, KxN) supplied as transposed matrix BT[n][k] row-major stride ldt:
//   lane half h, n=lane&15: 16 contiguous halves K=16h..16h+15 at column n.
__device__ __forceinline__ v16h load_b_frag(const _Float16* bt, int ldt) {
  const int lane = threadIdx.x & 31;
  const int hh = lane >> 4, n = lane & 15;
  return *(const v16h*)(bt + (size_t)n * ldt + 16 * hh);
}

__device__ __forceinline__ v8f wmma_f16(v16h a, v16h b, v8f c) {
  return __builtin_amdgcn_wmma_f32_16x16x32_f16(false, a, false, b, (short)0, c,
                                                false, false);
}

// gfx1250 async copy: 16 bytes global -> LDS per lane, tracked on ASYNCcnt.
__device__ __forceinline__ void async_b128(_Float16* lds, const _Float16* g) {
  unsigned lds_addr = (unsigned)(uintptr_t)lds;        // low 32 bits = LDS offset
  unsigned long long gaddr = (unsigned long long)(uintptr_t)g;
  asm volatile("global_load_async_to_lds_b128 %0, %1, off"
               :: "v"(lds_addr), "v"(gaddr) : "memory");
}

// ---- conversion kernels ---------------------------------------------------
__global__ void cvt_f32_to_f16(const float* __restrict__ src,
                               _Float16* __restrict__ dst, int n) {
  int i = blockIdx.x * blockDim.x + threadIdx.x;
  if (i < n) dst[i] = (_Float16)src[i];
}

// dst[b][c][r] = src[b][r][c]  (fp32 -> fp16 transpose)
__global__ void transpose_to_f16(const float* __restrict__ src,
                                 _Float16* __restrict__ dst,
                                 int batch, int rows, int cols) {
  int i = blockIdx.x * blockDim.x + threadIdx.x;
  int total = batch * rows * cols;
  if (i >= total) return;
  int c = i % cols;
  int r = (i / cols) % rows;
  int b = i / (cols * rows);
  dst[((size_t)b * cols + c) * rows + r] = (_Float16)src[i];
}

// ---- QKV projection: per wave a 32x64 tile of Q, K, V for one (b,h) -------
__global__ __launch_bounds__(256) void qkv_kernel(
    const _Float16* __restrict__ x16, const _Float16* __restrict__ wqT,
    const _Float16* __restrict__ wkT, const _Float16* __restrict__ wvT,
    _Float16* __restrict__ Q, _Float16* __restrict__ K,
    _Float16* __restrict__ VT) {
  const int tile = blockIdx.x * (blockDim.x >> 5) + (threadIdx.x >> 5);
  const int mt = tile % (kT / 32);
  const int bh = tile / (kT / 32);
  const int h = bh % kH, b = bh / kH;
  const int m0 = mt * 32;
  const int lane = threadIdx.x & 31;
  const int hh = lane >> 4, n = lane & 15;

  const _Float16* A = x16 + ((size_t)(b * kT + m0)) * kE;

#pragma unroll 1
  for (int which = 0; which < 3; ++which) {
    const _Float16* WT = (which == 0 ? wqT : which == 1 ? wkT : wvT) +
                         (size_t)h * kHS * kE;
    v8f acc[8] = {};  // [ms][nt]: 2 row-halves x 4 col tiles
#pragma unroll 1
    for (int k0 = 0; k0 < kE; k0 += 32) {
      v16h a0 = load_a_frag(A + k0, kE);
      v16h a1 = load_a_frag(A + (size_t)16 * kE + k0, kE);
#pragma unroll
      for (int t = 0; t < 4; ++t) {
        v16h bfr = load_b_frag(WT + (size_t)(t * 16) * kE + k0, kE);
        acc[t] = wmma_f16(a0, bfr, acc[t]);
        acc[4 + t] = wmma_f16(a1, bfr, acc[4 + t]);
      }
    }
    if (which < 2) {  // Q or K: [B][H][T][HS] row-major
      _Float16* out = (which == 0 ? Q : K) +
                      (((size_t)(b * kH + h) * kT + m0)) * kHS;
#pragma unroll
      for (int ms = 0; ms < 2; ++ms)
#pragma unroll
        for (int t = 0; t < 4; ++t)
#pragma unroll
          for (int r = 0; r < 8; ++r)
            out[(size_t)(ms * 16 + r + 8 * hh) * kHS + t * 16 + n] =
                (_Float16)acc[ms * 4 + t][r];
    } else {          // V transposed: [B][H][HS][T]
      _Float16* out = VT + ((size_t)(b * kH + h) * kHS) * kT;
#pragma unroll
      for (int ms = 0; ms < 2; ++ms)
#pragma unroll
        for (int t = 0; t < 4; ++t)
#pragma unroll
          for (int r = 0; r < 8; ++r)
            out[(size_t)(t * 16 + n) * kT + m0 + ms * 16 + r + 8 * hh] =
                (_Float16)acc[ms * 4 + t][r];
    }
  }
}

// ---- causal flash attention -----------------------------------------------
// Block = 8 waves covering 8 consecutive 16-row query tiles of one (b,h).
// K/V tiles are staged once per block into LDS with async-to-LDS copies.
__global__ __launch_bounds__(256) void flash_kernel(
    const _Float16* __restrict__ Q, const _Float16* __restrict__ K,
    const _Float16* __restrict__ VT, _Float16* __restrict__ O16) {
  __shared__ __align__(64) _Float16 Kl[32 * 64];   // 32 keys x 64 hs
  __shared__ __align__(64) _Float16 Vl[64 * 32];   // 64 hs x 32 keys (V^T)
  __shared__ __align__(64) _Float16 Pl[8][16 * 32];
  const int wid = threadIdx.x >> 5;
  const int g = blockIdx.x % (kRowTiles / 8);      // 16 tile-groups per (b,h)
  const int bh = blockIdx.x / (kRowTiles / 8);
  const int h = bh % kH, b = bh / kH;
  const int m0 = (g * 8 + wid) * 16;
  const int lane = threadIdx.x & 31;
  const int hh = lane >> 4, n = lane & 15;
  const int i = threadIdx.x;
  _Float16* pl = &Pl[wid][0];

  const _Float16* Qb = Q + (((size_t)(b * kH + h) * kT + m0)) * kHS;
  const _Float16* Kb = K + ((size_t)(b * kH + h) * kT) * kHS;
  const _Float16* Vb = VT + ((size_t)(b * kH + h) * kHS) * kT;

  const v16h qa0 = load_a_frag(Qb, kHS);       // hs 0..31
  const v16h qa1 = load_a_frag(Qb + 32, kHS);  // hs 32..63

  v8f acc[4] = {};
  float mrow[8], lrow[8];
#pragma unroll
  for (int r = 0; r < 8; ++r) { mrow[r] = -1e30f; lrow[r] = 0.f; }

  const float sscale = 0.125f * kLog2e;  // 1/sqrt(64) in log2 domain
  const int sEnd = m0 + 16;
  const int maxEnd = (g * 8 + 8) * 16;   // uniform across the block
  const int trow_base = m0 + 8 * hh;

  for (int s0 = 0; s0 < maxEnd; s0 += 32) {
    // cooperative stage: each of 256 threads copies one 16B chunk of K and V
    async_b128(Kl + (i >> 3) * 64 + (i & 7) * 8,
               Kb + (size_t)(s0 + (i >> 3)) * kHS + (i & 7) * 8);
    async_b128(Vl + (i >> 2) * 32 + (i & 3) * 8,
               Vb + (size_t)(i >> 2) * kT + s0 + (i & 3) * 8);
    asm volatile("s_wait_asynccnt 0" ::: "memory");
    __syncthreads();

    if (s0 < sEnd) {
      // scores: two 16x16 tiles (keys s0..+15 and s0+16..+31)
      v8f sc0 = {}, sc1 = {};
      sc0 = wmma_f16(qa0, load_b_frag(Kl, kHS), sc0);
      sc0 = wmma_f16(qa1, load_b_frag(Kl + 32, kHS), sc0);
      sc1 = wmma_f16(qa0, load_b_frag(Kl + 16 * kHS, kHS), sc1);
      sc1 = wmma_f16(qa1, load_b_frag(Kl + 16 * kHS + 32, kHS), sc1);

      // scale + causal mask (C layout: row = r + 8*hh, col = base + n)
#pragma unroll
      for (int r = 0; r < 8; ++r) {
        int t = trow_base + r;
        float sa = sc0[r] * sscale;
        float sb = sc1[r] * sscale;
        if (s0 + n > t) sa = -1e30f;
        if (s0 + 16 + n > t) sb = -1e30f;
        sc0[r] = sa; sc1[r] = sb;
      }

      // online softmax per row (reduce across the 16-lane half)
#pragma unroll
      for (int r = 0; r < 8; ++r) {
        float mx = fmaxf(sc0[r], sc1[r]);
#pragma unroll
        for (int off = 8; off >= 1; off >>= 1)
          mx = fmaxf(mx, __shfl_xor(mx, off, 32));
        float mnew = fmaxf(mrow[r], mx);
        float corr = __builtin_exp2f(mrow[r] - mnew);
        mrow[r] = mnew;
        float p0 = __builtin_exp2f(sc0[r] - mnew);
        float p1 = __builtin_exp2f(sc1[r] - mnew);
        sc0[r] = p0; sc1[r] = p1;
        float rs = p0 + p1;
#pragma unroll
        for (int off = 8; off >= 1; off >>= 1) rs += __shfl_xor(rs, off, 32);
        lrow[r] = lrow[r] * corr + rs;
#pragma unroll
        for (int t2 = 0; t2 < 4; ++t2) acc[t2][r] *= corr;
      }

      // C-layout P -> LDS (f16) -> A-layout fragment (wave-private tile)
#pragma unroll
      for (int r = 0; r < 8; ++r) {
        pl[(r + 8 * hh) * 32 + n] = (_Float16)sc0[r];
        pl[(r + 8 * hh) * 32 + 16 + n] = (_Float16)sc1[r];
      }
      asm volatile("s_wait_dscnt 0" ::: "memory");
      v16h pa = load_a_frag(pl, 32);
      asm volatile("s_wait_dscnt 0" ::: "memory");

      // P (16x32) x V (32x16 per hs-tile) from LDS V^T
#pragma unroll
      for (int t2 = 0; t2 < 4; ++t2) {
        v16h vb = load_b_frag(Vl + (t2 * 16) * 32, 32);
        acc[t2] = wmma_f16(pa, vb, acc[t2]);
      }
    }
    __syncthreads();
  }

  // normalize and store concat-head fp16 [B][T][H*HS]
  _Float16* out = O16 + ((size_t)(b * kT + m0)) * kC + h * kHS;
#pragma unroll
  for (int r = 0; r < 8; ++r) {
    float inv = 1.0f / lrow[r];
#pragma unroll
    for (int t2 = 0; t2 < 4; ++t2)
      out[(size_t)(r + 8 * hh) * kC + t2 * 16 + n] = (_Float16)(acc[t2][r] * inv);
  }
}

// ---- output projection: out[BT][E] = O16[BT][C] * wp[C][E], 32x64/wave ----
__global__ __launch_bounds__(256) void proj_kernel(
    const _Float16* __restrict__ O16, const _Float16* __restrict__ wpT,
    float* __restrict__ out) {
  const int tile = blockIdx.x * (blockDim.x >> 5) + (threadIdx.x >> 5);
  const int cg = tile % (kE / 64);
  const int mt = tile / (kE / 64);
  const int m0 = mt * 32, n0 = cg * 64;
  const int lane = threadIdx.x & 31;
  const int hh = lane >> 4, n = lane & 15;

  const _Float16* A = O16 + (size_t)m0 * kC;
  v8f acc[8] = {};
#pragma unroll 1
  for (int k0 = 0; k0 < kC; k0 += 32) {
    v16h a0 = load_a_frag(A + k0, kC);
    v16h a1 = load_a_frag(A + (size_t)16 * kC + k0, kC);
#pragma unroll
    for (int t = 0; t < 4; ++t) {
      v16h bfr = load_b_frag(wpT + (size_t)(n0 + t * 16) * kC + k0, kC);
      acc[t] = wmma_f16(a0, bfr, acc[t]);
      acc[4 + t] = wmma_f16(a1, bfr, acc[4 + t]);
    }
  }
#pragma unroll
  for (int ms = 0; ms < 2; ++ms)
#pragma unroll
    for (int t = 0; t < 4; ++t)
#pragma unroll
      for (int r = 0; r < 8; ++r)
        out[(size_t)(m0 + ms * 16 + r + 8 * hh) * kE + n0 + t * 16 + n] =
            acc[ms * 4 + t][r];
}

} // namespace

extern "C" void kernel_launch(void* const* d_in, const int* in_sizes, int n_in,
                              void* d_out, int out_size, void* d_ws,
                              size_t ws_size, hipStream_t stream) {
  (void)in_sizes; (void)n_in; (void)out_size; (void)ws_size;
  const float* x = (const float*)d_in[0];
  const float* wq = (const float*)d_in[1];
  const float* wk = (const float*)d_in[2];
  const float* wv = (const float*)d_in[3];
  const float* wp = (const float*)d_in[4];
  float* out = (float*)d_out;

  char* ws = (char*)d_ws;
  size_t off = 0;
  auto carve = [&](size_t bytes) {
    char* p = ws + off;
    off += (bytes + 255) & ~(size_t)255;
    return p;
  };
  _Float16* x16 = (_Float16*)carve((size_t)kB * kT * kE * 2);
  _Float16* q16 = (_Float16*)carve((size_t)kB * kH * kT * kHS * 2);
  _Float16* k16 = (_Float16*)carve((size_t)kB * kH * kT * kHS * 2);
  _Float16* vt16 = (_Float16*)carve((size_t)kB * kH * kT * kHS * 2);
  _Float16* o16 = (_Float16*)carve((size_t)kB * kT * kC * 2);
  _Float16* wqT = (_Float16*)carve((size_t)kH * kHS * kE * 2);
  _Float16* wkT = (_Float16*)carve((size_t)kH * kHS * kE * 2);
  _Float16* wvT = (_Float16*)carve((size_t)kH * kHS * kE * 2);
  _Float16* wpT = (_Float16*)carve((size_t)kC * kE * 2);

  const int nx = kB * kT * kE;
  cvt_f32_to_f16<<<(nx + 255) / 256, 256, 0, stream>>>(x, x16, nx);
  const int nw = kH * kE * kHS;
  transpose_to_f16<<<(nw + 255) / 256, 256, 0, stream>>>(wq, wqT, kH, kE, kHS);
  transpose_to_f16<<<(nw + 255) / 256, 256, 0, stream>>>(wk, wkT, kH, kE, kHS);
  transpose_to_f16<<<(nw + 255) / 256, 256, 0, stream>>>(wv, wvT, kH, kE, kHS);
  transpose_to_f16<<<(kC * kE + 255) / 256, 256, 0, stream>>>(wp, wpT, 1, kC, kE);

  const int qkv_tiles = kB * kH * (kT / 32);       // 2048 waves
  qkv_kernel<<<qkv_tiles / 8, 256, 0, stream>>>(x16, wqT, wkT, wvT, q16, k16,
                                                vt16);
  const int flash_blocks = kB * kH * (kRowTiles / 8);  // 512 blocks
  flash_kernel<<<flash_blocks, 256, 0, stream>>>(q16, k16, vt16, o16);
  const int ptiles = (kB * kT / 32) * (kE / 64);   // 2048 waves
  proj_kernel<<<ptiles / 8, 256, 0, stream>>>(o16, wpT, out);
}
